// MultiHeadAttention_32693291057807
// MI455X (gfx1250) — matrix operand order repaired and checked by reading, not compiled, not run on hardware
//
#include <hip/hip_runtime.h>
#include <hip/hip_bf16.h>

// ---------------------------------------------------------------------------
// MI455X (gfx1250) multi-head attention, f16 WMMA pipeline.
// B=4, T=1024, H=16, Hs=64, C=1024.  ~52 GFLOP, ~50MB working set (fits L2).
// All GEMMs via v_wmma_f32_16x16x32_f16.  Per-wave tiles use the
// "1 shared A fragment + few named accumulators" shape that the register
// allocator coalesces cleanly (in-place WMMA accumulation, no v_mov storms).
// ---------------------------------------------------------------------------

typedef __attribute__((ext_vector_type(16))) _Float16 v16h;
typedef __attribute__((ext_vector_type(8)))  _Float16 v8h;
typedef __attribute__((ext_vector_type(8)))  float    v8f;

#define N_HEADS   16
#define HEAD_SIZE 64
#define D_EMBD    1024
#define BATCH     4
#define SEQ       1024
#define NROWS     (BATCH * SEQ)          // 4096

__device__ __forceinline__ v16h make_v16(v8h lo, v8h hi) {
  return __builtin_shufflevector(lo, hi, 0,1,2,3,4,5,6,7,8,9,10,11,12,13,14,15);
}

__device__ __forceinline__ v8f wmma_f16(v16h a, v16h b, v8f c) {
  return __builtin_amdgcn_wmma_f32_16x16x32_f16(
      /*neg_a=*/false, a, /*neg_b=*/false, b,
      /*c_mod=*/(short)0, c, /*reuse_a=*/false, /*reuse_b=*/false);
}

// A fragment (16x32 f16, M x K) from row-major [M][lda] starting at `row0`.
// Lane m = lane&15, half = lane>>4:
//   elem j -> K = k0 + half*16 + j
__device__ __forceinline__ v16h load_a(const _Float16* row0, int lda, int k0, int lane) {
  const int m = lane & 15, half = (lane >> 4) & 1;
  const _Float16* p = row0 + (size_t)m * lda + k0 + half * 16;
  v8h lo = *(const v8h*)(p);
  v8h hi = *(const v8h*)(p + 8);
  return make_v16(lo, hi);
}

// B fragment (32x16 f16, K x N) from N-major storage [N][ldb]:
//   lane n = lane&15, half = lane>>4, elem j -> K = k0 + half*16 + j
__device__ __forceinline__ v16h load_b(const _Float16* col0, int ldb, int k0, int lane) {
  const int n = lane & 15, half = (lane >> 4) & 1;
  return *(const v16h*)(col0 + (size_t)n * ldb + k0 + half * 16);
}

// -------------------------- prep kernels -----------------------------------

__global__ void cvt_f32_f16_kernel(const float* __restrict__ in,
                                   _Float16* __restrict__ out, int n) {
  int i = blockIdx.x * blockDim.x + threadIdx.x;
  if (i < n) out[i] = (_Float16)in[i];
}

// W [16][1024][64] f32  ->  Wt [16][64][1024] f16   (so B frags are contiguous)
__global__ void transpose_w_kernel(const float* __restrict__ W,
                                   _Float16* __restrict__ Wt) {
  int i = blockIdx.x * blockDim.x + threadIdx.x;   // 16*1024*64 = 1048576
  int h = i >> 16;
  int c = (i >> 6) & 1023;
  int s = i & 63;
  Wt[((size_t)((h << 6) + s)) * D_EMBD + c] = (_Float16)W[i];
}

// -------------------------- fused QKV projection ---------------------------
// 16x16 tile per wave, A fragment reused across Q/K/V (3 WMMA / K-step).
// Writes: Qh,Kh as [bh][t][s] f16 ; V transposed as Vt [bh][s][t] f16.
__global__ void qkv_kernel(const _Float16* __restrict__ xh,
                           const _Float16* __restrict__ wqt,
                           const _Float16* __restrict__ wkt,
                           const _Float16* __restrict__ wvt,
                           _Float16* __restrict__ Qh,
                           _Float16* __restrict__ Kh,
                           _Float16* __restrict__ Vt) {
  const int lane = threadIdx.x & 31;
  const int wave = blockIdx.x * (blockDim.x >> 5) + (threadIdx.x >> 5);
  const int mt = wave >> 6;          // 0..255  (M = 4096)
  const int nt = wave & 63;          // 0..63   (N = 1024)
  const int m0 = mt * 16, n0 = nt * 16;

  const _Float16* arow = xh  + (size_t)m0 * D_EMBD;
  const _Float16* bq   = wqt + (size_t)n0 * D_EMBD;
  const _Float16* bk   = wkt + (size_t)n0 * D_EMBD;
  const _Float16* bv   = wvt + (size_t)n0 * D_EMBD;

  v8f cq = {}, ck = {}, cv = {};
  for (int k0 = 0; k0 < D_EMBD; k0 += 32) {
    v16h a   = load_a(arow, D_EMBD, k0, lane);
    v16h bqf = load_b(bq, D_EMBD, k0, lane);
    v16h bkf = load_b(bk, D_EMBD, k0, lane);
    v16h bvf = load_b(bv, D_EMBD, k0, lane);
    cq = wmma_f16(a, bqf, cq);
    ck = wmma_f16(a, bkf, ck);
    cv = wmma_f16(a, bvf, cv);
  }

  const int half = lane >> 4, col = lane & 15;
  const int b = m0 >> 10;            // batch
  const int h = n0 >> 6;             // head
  const int s = (n0 & 63) + col;     // head-dim column
  const int bh = b * N_HEADS + h;
#pragma unroll
  for (int r = 0; r < 8; ++r) {
    const int t = (m0 & 1023) + r + half * 8;
    const size_t qi = (((size_t)bh * SEQ) + t) * HEAD_SIZE + s;
    Qh[qi] = (_Float16)cq[r];
    Kh[qi] = (_Float16)ck[r];
    const size_t vi = (((size_t)bh * HEAD_SIZE) + s) * SEQ + t;
    Vt[vi] = (_Float16)cv[r];
  }
}

// -------------------------- flash attention --------------------------------
// One wave per 16-row Q tile of one (b,h).  Online softmax over 32-key chunks.
__global__ void attn_kernel(const _Float16* __restrict__ Qh,
                            const _Float16* __restrict__ Kh,
                            const _Float16* __restrict__ Vt,
                            _Float16* __restrict__ concat) {
  __shared__ __align__(32) _Float16 plds[8][16 * 32];   // 1KB / wave

  const int lane  = threadIdx.x & 31;
  const int wslot = threadIdx.x >> 5;
  const int wave  = blockIdx.x * 8 + wslot;
  const int qt = wave & 63;                // q-tile within sequence (T/16 = 64)
  const int bh = wave >> 6;                // 0..63

  const _Float16* Qbh = Qh + (size_t)bh * SEQ * HEAD_SIZE;
  const _Float16* Kbh = Kh + (size_t)bh * SEQ * HEAD_SIZE;
  const _Float16* Vbh = Vt + (size_t)bh * HEAD_SIZE * SEQ;

  const int m = lane & 15, half = lane >> 4, col = m;

  // Q tile: A fragments for K-dim s in [0,32) and [32,64)
  const _Float16* qrow = Qbh + (size_t)(qt * 16 + m) * HEAD_SIZE;
  v16h a0 = make_v16(*(const v8h*)(qrow + half * 16),
                     *(const v8h*)(qrow + 8 + half * 16));
  v16h a1 = make_v16(*(const v8h*)(qrow + 32 + half * 16),
                     *(const v8h*)(qrow + 40 + half * 16));

  v8f O0 = {}, O1 = {}, O2 = {}, O3 = {};
  float mrow[8], lrow[8];
#pragma unroll
  for (int r = 0; r < 8; ++r) { mrow[r] = -1e30f; lrow[r] = 0.0f; }

  const float scale = 0.125f;              // 1/sqrt(64)
  _Float16* pl = &plds[wslot][0];
  const int nchunks = (qt >> 1) + 1;       // 32-key chunks up to the diagonal

  for (int c = 0; c < nchunks; ++c) {
    const int u0 = c * 32;
    // Scores: S0 covers keys u0..u0+15, S1 covers u0+16..u0+31
    const _Float16* k0p = Kbh + (size_t)(u0 + col) * HEAD_SIZE;
    const _Float16* k1p = Kbh + (size_t)(u0 + 16 + col) * HEAD_SIZE;
    v16h kb00 = *(const v16h*)(k0p + half * 16);
    v16h kb01 = *(const v16h*)(k0p + 32 + half * 16);
    v16h kb10 = *(const v16h*)(k1p + half * 16);
    v16h kb11 = *(const v16h*)(k1p + 32 + half * 16);
    v8f S0 = {}, S1 = {};
    S0 = wmma_f16(a0, kb00, S0);
    S0 = wmma_f16(a1, kb01, S0);
    S1 = wmma_f16(a0, kb10, S1);
    S1 = wmma_f16(a1, kb11, S1);

    // Online softmax over the 32 new columns, rescale accumulators,
    // stash P in LDS (f16, [row][u_local]) for the A-fragment of P@V.
#pragma unroll
    for (int r = 0; r < 8; ++r) {
      const int t = qt * 16 + r + half * 8;
      float s0v = S0[r] * scale; if (u0 + col      > t) s0v = -1e30f;
      float s1v = S1[r] * scale; if (u0 + 16 + col > t) s1v = -1e30f;
      float mx = fmaxf(s0v, s1v);
#pragma unroll
      for (int off = 1; off < 16; off <<= 1)
        mx = fmaxf(mx, __shfl_xor(mx, off, 16));
      const float mnew = fmaxf(mrow[r], mx);
      const float corr = __expf(mrow[r] - mnew);
      const float p0 = __expf(s0v - mnew);
      const float p1 = __expf(s1v - mnew);
      float rs = p0 + p1;
#pragma unroll
      for (int off = 1; off < 16; off <<= 1)
        rs += __shfl_xor(rs, off, 16);
      lrow[r] = lrow[r] * corr + rs;
      mrow[r] = mnew;
      O0[r] *= corr; O1[r] *= corr; O2[r] *= corr; O3[r] *= corr;
      const int row = r + half * 8;
      pl[row * 32 + col]      = (_Float16)p0;
      pl[row * 32 + 16 + col] = (_Float16)p1;
    }

    asm volatile("s_wait_dscnt 0" ::: "memory");   // LDS RAW across lanes

    // P as A fragment (16x32)
    const _Float16* lp = pl + m * 32 + half * 16;
    v16h pa = make_v16(*(const v8h*)(lp), *(const v8h*)(lp + 8));

    // V B-fragments: Vt is [s][t], so lane n reads 16 contiguous keys.
    v16h vb0 = *(const v16h*)(Vbh + (size_t)( 0 + col) * SEQ + u0 + half * 16);
    v16h vb1 = *(const v16h*)(Vbh + (size_t)(16 + col) * SEQ + u0 + half * 16);
    v16h vb2 = *(const v16h*)(Vbh + (size_t)(32 + col) * SEQ + u0 + half * 16);
    v16h vb3 = *(const v16h*)(Vbh + (size_t)(48 + col) * SEQ + u0 + half * 16);
    O0 = wmma_f16(pa, vb0, O0);
    O1 = wmma_f16(pa, vb1, O1);
    O2 = wmma_f16(pa, vb2, O2);
    O3 = wmma_f16(pa, vb3, O3);

    asm volatile("s_wait_dscnt 0" ::: "memory");   // before next chunk rewrites pl
  }

  // Normalize and store concat[b][t][h*64+s] (f16 input to out-proj GEMM)
  const int b = bh >> 4, h = bh & 15;
#pragma unroll
  for (int r = 0; r < 8; ++r) {
    const float inv = 1.0f / lrow[r];
    const int t = qt * 16 + r + half * 8;
    const size_t base = ((size_t)(b * SEQ + t)) * D_EMBD + h * HEAD_SIZE;
    concat[base +  0 + col] = (_Float16)(O0[r] * inv);
    concat[base + 16 + col] = (_Float16)(O1[r] * inv);
    concat[base + 32 + col] = (_Float16)(O2[r] * inv);
    concat[base + 48 + col] = (_Float16)(O3[r] * inv);
  }
}

// -------------------------- output projection ------------------------------
// out[n][d] = sum_c concat[n][c] * Wo[d][c] + bo[d]   (Wo row-major == N-major B)
// 16x64 tile per wave: 1 shared A fragment, 4 B fragments, 4 named
// accumulators -> 4 WMMA per K-step (same allocator-friendly shape as qkv).
__global__ void oproj_kernel(const _Float16* __restrict__ concat,
                             const _Float16* __restrict__ woh,
                             const float* __restrict__ bo,
                             float* __restrict__ out) {
  const int lane = threadIdx.x & 31;
  const int wave = blockIdx.x * (blockDim.x >> 5) + (threadIdx.x >> 5);
  const int mt = wave >> 4;          // 0..255  (M = 4096)
  const int nt = wave & 15;          // 0..15   (N = 1024, 64-col blocks)
  const int m0 = mt * 16, n0 = nt * 64;

  const _Float16* arow = concat + (size_t)m0 * D_EMBD;
  const _Float16* b0p = woh + (size_t)(n0)      * D_EMBD;
  const _Float16* b1p = woh + (size_t)(n0 + 16) * D_EMBD;
  const _Float16* b2p = woh + (size_t)(n0 + 32) * D_EMBD;
  const _Float16* b3p = woh + (size_t)(n0 + 48) * D_EMBD;

  v8f c0 = {}, c1 = {}, c2 = {}, c3 = {};
  for (int k0 = 0; k0 < D_EMBD; k0 += 32) {
    v16h a  = load_a(arow, D_EMBD, k0, lane);
    v16h f0 = load_b(b0p, D_EMBD, k0, lane);
    v16h f1 = load_b(b1p, D_EMBD, k0, lane);
    v16h f2 = load_b(b2p, D_EMBD, k0, lane);
    v16h f3 = load_b(b3p, D_EMBD, k0, lane);
    c0 = wmma_f16(a, f0, c0);
    c1 = wmma_f16(a, f1, c1);
    c2 = wmma_f16(a, f2, c2);
    c3 = wmma_f16(a, f3, c3);
  }

  const int half = lane >> 4, col = lane & 15;
  const float bias0 = bo[n0 + col];
  const float bias1 = bo[n0 + 16 + col];
  const float bias2 = bo[n0 + 32 + col];
  const float bias3 = bo[n0 + 48 + col];
#pragma unroll
  for (int r = 0; r < 8; ++r) {
    const size_t row = (size_t)(m0 + r + half * 8) * D_EMBD;
    out[row + n0 + col]      = c0[r] + bias0;
    out[row + n0 + 16 + col] = c1[r] + bias1;
    out[row + n0 + 32 + col] = c2[r] + bias2;
    out[row + n0 + 48 + col] = c3[r] + bias3;
  }
}

// -------------------------- host launcher ----------------------------------

extern "C" void kernel_launch(void* const* d_in, const int* in_sizes, int n_in,
                              void* d_out, int out_size, void* d_ws, size_t ws_size,
                              hipStream_t stream) {
  const float* x  = (const float*)d_in[0];   // [4,1024,1024]
  const float* Wq = (const float*)d_in[1];   // [16,1024,64]
  const float* Wk = (const float*)d_in[2];
  const float* Wv = (const float*)d_in[3];
  const float* Wo = (const float*)d_in[4];   // [1024,1024]
  const float* bo = (const float*)d_in[5];   // [1024]
  float* out = (float*)d_out;

  char* ws = (char*)d_ws;
  const size_t SZ_XH   = (size_t)NROWS * D_EMBD * 2;            // 8 MiB
  const size_t SZ_W    = (size_t)D_EMBD * D_EMBD * 2;           // 2 MiB
  const size_t SZ_QKV  = (size_t)BATCH * N_HEADS * SEQ * HEAD_SIZE * 2; // 8 MiB
  _Float16* xh   = (_Float16*)(ws);
  _Float16* wqt  = (_Float16*)(ws + SZ_XH);
  _Float16* wkt  = (_Float16*)(ws + SZ_XH + SZ_W);
  _Float16* wvt  = (_Float16*)(ws + SZ_XH + 2 * SZ_W);
  _Float16* woh  = (_Float16*)(ws + SZ_XH + 3 * SZ_W);
  _Float16* Qh   = (_Float16*)(ws + SZ_XH + 4 * SZ_W);
  _Float16* Kh   = (_Float16*)(ws + SZ_XH + 4 * SZ_W + SZ_QKV);
  _Float16* Vt   = (_Float16*)(ws + SZ_XH + 4 * SZ_W + 2 * SZ_QKV);
  _Float16* conc = (_Float16*)(ws + SZ_XH + 4 * SZ_W + 3 * SZ_QKV);
  // total = 8 + 4*2 + 4*8 = 48 MiB

  // Stage 0: precision conversion + weight transposes
  cvt_f32_f16_kernel<<<(NROWS * D_EMBD) / 256, 256, 0, stream>>>(x, xh, NROWS * D_EMBD);
  transpose_w_kernel<<<(N_HEADS * D_EMBD * HEAD_SIZE) / 256, 256, 0, stream>>>(Wq, wqt);
  transpose_w_kernel<<<(N_HEADS * D_EMBD * HEAD_SIZE) / 256, 256, 0, stream>>>(Wk, wkt);
  transpose_w_kernel<<<(N_HEADS * D_EMBD * HEAD_SIZE) / 256, 256, 0, stream>>>(Wv, wvt);
  cvt_f32_f16_kernel<<<(D_EMBD * D_EMBD) / 256, 256, 0, stream>>>(Wo, woh, D_EMBD * D_EMBD);

  // Stage 1: fused QKV projection (16384 tile-waves, 8 waves/block)
  qkv_kernel<<<2048, 256, 0, stream>>>(xh, wqt, wkt, wvt, Qh, Kh, Vt);

  // Stage 2: flash attention (4096 q-tile waves)
  attn_kernel<<<512, 256, 0, stream>>>(Qh, Kh, Vt, conc);

  // Stage 3: output projection + bias (4096 tile-waves, 16x64 tiles)
  oproj_kernel<<<512, 256, 0, stream>>>(conc, woh, bo, out);
}
